// GAT_30897994727614
// MI455X (gfx1250) — matrix operand order repaired and checked by reading, not compiled, not run on hardware
//
#include <hip/hip_runtime.h>
#include <hip/hip_bf16.h>
#include <stdint.h>

typedef __attribute__((ext_vector_type(16))) __bf16 v16bf;
typedef __attribute__((ext_vector_type(8)))  float  v8f;
typedef __attribute__((ext_vector_type(4)))  int    i4v;
typedef __attribute__((ext_vector_type(4)))  float  f4v;
typedef __attribute__((ext_vector_type(8)))  int    i8v;

#define NN     8192
#define INF_   256
#define OUTF   128
#define JSPLIT 4

__device__ __forceinline__ float lrelu(float x) { return x > 0.f ? x : 0.01f * x; }

// ---------------------------------------------------------------------------
// K1: H = x @ W + bias  via v_wmma_f32_16x16x32_bf16.
// A layout (16-bit, 16x32): lane = M (mod 16); slot s -> K = (s<8?s:s+8)+half*8
// B layout (16-bit, 32x16): lane = N (mod 16); slot s -> K = (s<8?s:s+8)+half*8
// ---------------------------------------------------------------------------
__global__ void __launch_bounds__(128) k_gemm_H(
    const float* __restrict__ x, const float* __restrict__ w,
    const float* __restrict__ bias, float* __restrict__ H)
{
  const int lane = threadIdx.x & 31, wid = threadIdx.x >> 5;
  const int tile = blockIdx.x * 4 + wid;
  const int i0 = (tile >> 3) * 16;       // 512 i-tiles
  const int n0 = (tile & 7) * 16;        // 8 n-tiles
  const int M = lane & 15, half = lane >> 4;

  const float* xr = x + (size_t)(i0 + M) * INF_;
  v8f acc = {};
  #pragma unroll
  for (int k0 = 0; k0 < INF_; k0 += 32) {
    float av[16];
    *(f4v*)(av + 0)  = *(const f4v*)(xr + k0 + half * 8 + 0);
    *(f4v*)(av + 4)  = *(const f4v*)(xr + k0 + half * 8 + 4);
    *(f4v*)(av + 8)  = *(const f4v*)(xr + k0 + 16 + half * 8 + 0);
    *(f4v*)(av + 12) = *(const f4v*)(xr + k0 + 16 + half * 8 + 4);
    v16bf afrag, bfrag;
    #pragma unroll
    for (int s = 0; s < 16; ++s) afrag[s] = (__bf16)av[s];
    #pragma unroll
    for (int s = 0; s < 16; ++s)
      bfrag[s] = (__bf16)w[(size_t)(k0 + (s < 8 ? s : s + 8) + half * 8) * OUTF + n0 + M];
    acc = __builtin_amdgcn_wmma_f32_16x16x32_bf16(false, afrag, false, bfrag,
                                                  (short)0, acc, false, false);
  }
  const float bv = bias[n0 + M];
  #pragma unroll
  for (int k = 0; k < 8; ++k)   // C layout: VGPR k -> row M=k+8*half, col = lane%16
    H[(size_t)(i0 + k + half * 8) * OUTF + n0 + M] = acc[k] + bv;
}

// ---------------------------------------------------------------------------
// K1b: one wave per row: h1 = H[i].phi1, h2 = H[i].phi2, and write H (bf16)
// into a pre-swizzled global layout == the exact WMMA B-fragment byte layout,
// so the main kernel can async-DMA tiles straight into LDS.
// ---------------------------------------------------------------------------
__global__ void __launch_bounds__(256) k_rows(
    const float* __restrict__ H, const float* __restrict__ phi,
    __bf16* __restrict__ hswz, float* __restrict__ h1, float* __restrict__ h2)
{
  const int lane = threadIdx.x & 31, wid = threadIdx.x >> 5;
  const int row = blockIdx.x * 8 + wid;
  const float* hr = H + (size_t)row * OUTF;
  const int jh = (row >> 3) & 1, s = (row & 7) | ((row >> 1) & 8);
  __bf16* tdst = hswz + (size_t)(row >> 5) * 4096;
  float p1 = 0.f, p2 = 0.f;
  #pragma unroll
  for (int c = 0; c < 4; ++c) {
    const int d = lane + 32 * c;
    const float v = hr[d];
    p1 += v * phi[d];
    p2 += v * phi[OUTF + d];
    tdst[((((d >> 4) * 16) + (d & 15)) * 2 + jh) * 16 + s] = (__bf16)v;
  }
  #pragma unroll
  for (int o = 16; o > 0; o >>= 1) {
    p1 += __shfl_down(p1, o, 32);
    p2 += __shfl_down(p2, o, 32);
  }
  if (lane == 0) { h1[row] = p1; h2[row] = p2; }
}

// K1c: global max of h2 -> softmax stabilizer upper bound
__global__ void __launch_bounds__(256) k_maxh2(const float* __restrict__ h2,
                                               float* __restrict__ mx)
{
  __shared__ float sm[256];
  float m = -3.4e38f;
  for (int i = threadIdx.x; i < NN; i += 256) m = fmaxf(m, h2[i]);
  sm[threadIdx.x] = m;
  __syncthreads();
  for (int o = 128; o > 0; o >>= 1) {
    if (threadIdx.x < o) sm[threadIdx.x] = fmaxf(sm[threadIdx.x], sm[threadIdx.x + o]);
    __syncthreads();
  }
  if (threadIdx.x == 0) mx[0] = sm[0];
}

// ---------------------------------------------------------------------------
// K2: fused masked-softmax + aggregation. Fixed stabilizer
// c_i = lrelu(h1[i] + max h2) upper-bounds every row max -> no rescaling,
// partial sums over j are plain adds (JSPLIT-way parallel).
// Double-buffered async DMA: tile t+1 streams into LDS while tile t's
// scores (adj int4 + exp) and 8 bf16 WMMAs execute. adj/h2 register loads
// are software-pipelined one iteration ahead.
// ---------------------------------------------------------------------------
__global__ void __launch_bounds__(64) k_attn(
    const int* __restrict__ adj, const __bf16* __restrict__ hswz,
    const float* __restrict__ h1, const float* __restrict__ h2,
    const float* __restrict__ mxp,
    float* __restrict__ nump, float* __restrict__ zp)
{
  __shared__ __align__(16) char ldsB[2][8192];
  const int tid = threadIdx.x;
  const int lane = tid & 31, wid = tid >> 5;
  const int ib = blockIdx.x / JSPLIT, js = blockIdx.x % JSPLIT;
  const int i0 = ib * 32 + wid * 16;
  const int M = lane & 15, half = lane >> 4;
  const int i = i0 + M;

  const float h1i = h1[i];
  const float ci  = lrelu(h1i + mxp[0]);
  const int jstart = js * (NN / JSPLIT);
  const int NT = (NN / JSPLIT) / 32;            // 64 tiles of 32 j's
  const int* arow = adj + (size_t)i * NN;
  const uint32_t lb0 = (uint32_t)(uintptr_t)(&ldsB[0][0]) + (uint32_t)tid * 128u;

  // DMA one 8KB pre-swizzled bf16 H tile into LDS buffer `buf`.
  // 8 async instructions per wave, each instr moves 32 lanes x 16B = 512B.
  auto issue_dma = [&](int buf, int j0) {
    const uint32_t lb = lb0 + (uint32_t)buf * 8192u;
    const char* gs = (const char*)(hswz + (size_t)(j0 >> 5) * 4096) + (size_t)tid * 128;
    #pragma unroll
    for (int q = 0; q < 8; ++q) {
      asm volatile("global_load_async_to_lds_b128 %0, %1, off"
                   :: "v"(lb + (uint32_t)(q * 16)),
                      "v"((uint64_t)(uintptr_t)(gs + q * 16))
                   : "memory");
    }
  };
  // Per-lane adj + h2 values for the 16 A-fragment slots of one 32-j tile.
  // slot s -> K = (s<8 ? s : s+8) + half*8 : two contiguous runs of 8.
  auto load_regs = [&](int j0, int* va, float* hv) {
    *(i4v*)(va + 0)  = *(const i4v*)(arow + j0 + half * 8 + 0);
    *(i4v*)(va + 4)  = *(const i4v*)(arow + j0 + half * 8 + 4);
    *(i4v*)(va + 8)  = *(const i4v*)(arow + j0 + 16 + half * 8 + 0);
    *(i4v*)(va + 12) = *(const i4v*)(arow + j0 + 16 + half * 8 + 4);
    *(f4v*)(hv + 0)  = *(const f4v*)(h2 + j0 + half * 8 + 0);
    *(f4v*)(hv + 4)  = *(const f4v*)(h2 + j0 + half * 8 + 4);
    *(f4v*)(hv + 8)  = *(const f4v*)(h2 + j0 + 16 + half * 8 + 0);
    *(f4v*)(hv + 12) = *(const f4v*)(h2 + j0 + 16 + half * 8 + 4);
  };

  v8f acc[8] = {};
  float zsum = 0.f;
  int va[16];
  float hv[16];

  load_regs(jstart, va, hv);       // prologue: tile 0 regs + DMA
  issue_dma(0, jstart);

  for (int t = 0; t < NT; ++t) {
    const int j0 = jstart + t * 32;

    // ---- exp-weights for tile t (regs loaded an iteration ago) ----
    v16bf afrag;
    #pragma unroll
    for (int s = 0; s < 16; ++s) {
      const int K = (s < 8 ? s : s + 8) + half * 8;
      const float sc  = lrelu(h1i + hv[s]);
      const bool keep = (va[s] != 0) || ((j0 + K) == i);
      const float wgt = keep ? __expf(sc - ci) : 0.f;   // masked -> exact 0
      zsum += wgt;
      afrag[s] = (__bf16)wgt;
    }

    // ---- pipeline tile t+1: DMA into the other buffer + reg prefetch ----
    if (t + 1 < NT) {
      issue_dma((t + 1) & 1, j0 + 32);
      load_regs(j0 + 32, va, hv);
      asm volatile("s_wait_asynccnt 0x8" ::: "memory");  // tile t's 8 done
    } else {
      asm volatile("s_wait_asynccnt 0x0" ::: "memory");
    }
    __syncthreads();

    // ---- 8 WMMAs: (16x32 P) x (32x16 H-chunk) per 16-col group ----
    const char* lb = &ldsB[t & 1][0];
    #pragma unroll
    for (int u = 0; u < 8; ++u) {
      const i8v braw = *(const i8v*)(lb + (u * 16 + M) * 64 + half * 32);
      const v16bf bfrag = __builtin_bit_cast(v16bf, braw);
      acc[u] = __builtin_amdgcn_wmma_f32_16x16x32_bf16(false, afrag, false, bfrag,
                                                       (short)0, acc[u], false, false);
    }
    __syncthreads();   // all waves done reading buf before it's re-filled
  }

  // Z partial: lanes M and M+16 each hold half of row i's sum
  zsum += __shfl_xor(zsum, 16, 32);
  if (lane < 16) zp[(size_t)js * NN + i] = zsum;

  // numerator partials, C layout: VGPR k -> row k+8*half, col = u*16 + lane%16
  float* np = nump + (size_t)js * NN * OUTF;
  #pragma unroll
  for (int u = 0; u < 8; ++u)
    #pragma unroll
    for (int k = 0; k < 8; ++k)
      np[(size_t)(i0 + k + half * 8) * OUTF + u * 16 + M] = acc[u][k];
}

// K3: combine JSPLIT partials and normalize
__global__ void __launch_bounds__(256) k_final(
    const float* __restrict__ nump, const float* __restrict__ zp,
    float* __restrict__ out)
{
  const int idx = blockIdx.x * 256 + threadIdx.x;
  const int row = idx >> 7;   // OUTF == 128
  const int col = idx & 127;
  float z = 0.f, s = 0.f;
  #pragma unroll
  for (int j = 0; j < JSPLIT; ++j) {
    z += zp[(size_t)j * NN + row];
    s += nump[((size_t)j * NN + row) * OUTF + col];
  }
  out[idx] = s / z;
}

extern "C" void kernel_launch(void* const* d_in, const int* in_sizes, int n_in,
                              void* d_out, int out_size, void* d_ws, size_t ws_size,
                              hipStream_t stream)
{
  const int*   adj  = (const int*)d_in[0];
  const float* x    = (const float*)d_in[1];
  const float* w    = (const float*)d_in[2];
  const float* bias = (const float*)d_in[3];
  const float* phi  = (const float*)d_in[4];
  float* out = (float*)d_out;
  (void)in_sizes; (void)n_in; (void)out_size; (void)ws_size;

  char* ws = (char*)d_ws;
  size_t off = 0;
  auto take = [&](size_t bytes) -> char* {
    char* p = ws + off;
    off = (off + bytes + 255) & ~(size_t)255;
    return p;
  };
  float*  H    = (float*) take((size_t)NN * OUTF * 4);          // 4 MB
  __bf16* hswz = (__bf16*)take((size_t)NN * OUTF * 2);          // 2 MB swizzled
  float*  h1   = (float*) take((size_t)NN * 4);
  float*  h2   = (float*) take((size_t)NN * 4);
  float*  mx   = (float*) take(256);
  float*  zp   = (float*) take((size_t)JSPLIT * NN * 4);
  float*  nump = (float*) take((size_t)JSPLIT * NN * OUTF * 4); // 16 MB

  k_gemm_H<<<(NN / 16) * (OUTF / 16) / 4, 128, 0, stream>>>(x, w, bias, H);
  k_rows  <<<NN / 8, 256, 0, stream>>>(H, phi, hswz, h1, h2);
  k_maxh2 <<<1, 256, 0, stream>>>(h2, mx);
  k_attn  <<<(NN / 32) * JSPLIT, 64, 0, stream>>>(adj, hswz, h1, h2, mx, nump, zp);
  k_final <<<(NN * OUTF) / 256, 256, 0, stream>>>(nump, zp, out);
}
